// ImprovedGNN_12893491822683
// MI455X (gfx1250) — matrix operand order repaired and checked by reading, not compiled, not run on hardware
//
#include <hip/hip_runtime.h>
#include <hip/hip_bf16.h>

// ---------------------------------------------------------------------------
// Problem constants (match reference)
// ---------------------------------------------------------------------------
#define NN  50000
#define EE  800000
#define DD  128
#define HH  128
#define OO  64
#define LL  3

typedef __attribute__((ext_vector_type(16))) __bf16 v16bf;
typedef __attribute__((ext_vector_type(8)))  __bf16 v8bf;
typedef __attribute__((ext_vector_type(8)))  float  v8f;

// ---------------------------------------------------------------------------
// Helpers
// ---------------------------------------------------------------------------
__device__ __forceinline__ unsigned short f2bf(float f) {
    unsigned int u = __float_as_uint(f);
    // round-to-nearest-even bf16 truncation
    unsigned int r = (u + 0x7FFFu + ((u >> 16) & 1u)) >> 16;
    return (unsigned short)r;
}

__device__ __forceinline__ float gelu_exact(float x) {
    return 0.5f * x * (1.0f + erff(x * 0.70710678118654752440f));
}

// A-operand tile (16x32 bf16): lane l holds row m=l%16; two contiguous
// 16-byte chunks at k-offsets (l/16)*8 and (l/16)*8 + 16.
__device__ __forceinline__ v16bf load_a_tile(const unsigned short* p) {
    v8bf lo = *reinterpret_cast<const v8bf*>(p);
    v8bf hi = *reinterpret_cast<const v8bf*>(p + 16);
    v16bf a;
#pragma unroll
    for (int i = 0; i < 8; ++i) { a[i] = lo[i]; a[i + 8] = hi[i]; }
    return a;
}

// B-operand tile (32x16 bf16): lane l holds column n=l%16; k-rows (l/16)*8+i and (l/16)*8+16+i (stride ld).
__device__ __forceinline__ v16bf load_b_tile(const unsigned short* p, int ld) {
    v16bf b;
#pragma unroll
    for (int i = 0; i < 8; ++i) {
        b[i] = __builtin_bit_cast(__bf16, p[(size_t)i * ld]);
        b[i + 8] = __builtin_bit_cast(__bf16, p[(size_t)(i + 16) * ld]);
    }
    return b;
}

// ---------------------------------------------------------------------------
// fp32 -> bf16 conversion (weights)
// ---------------------------------------------------------------------------
__global__ void cvt_bf16_kernel(const float* __restrict__ in,
                                unsigned short* __restrict__ out, int n) {
    int i = blockIdx.x * blockDim.x + threadIdx.x;
    if (i < n) out[i] = f2bf(in[i]);
}

// ---------------------------------------------------------------------------
// Row layernorm (128 wide) -> bf16 output. One thread per node.
// ---------------------------------------------------------------------------
__global__ void ln_to_bf16_kernel(const float* __restrict__ x,
                                  const float* __restrict__ g,
                                  const float* __restrict__ b,
                                  unsigned short* __restrict__ out) {
    int node = blockIdx.x * blockDim.x + threadIdx.x;
    if (node >= NN) return;
    const float4* row = reinterpret_cast<const float4*>(x + (size_t)node * HH);
    float s = 0.f, ss = 0.f;
#pragma unroll 8
    for (int i = 0; i < HH / 4; ++i) {
        float4 v = row[i];
        s  += v.x + v.y + v.z + v.w;
        ss += v.x * v.x + v.y * v.y + v.z * v.z + v.w * v.w;
    }
    float m   = s * (1.0f / HH);
    float var = ss * (1.0f / HH) - m * m;
    float inv = rsqrtf(var + 1e-5f);
    unsigned short* o = out + (size_t)node * HH;
#pragma unroll 8
    for (int i = 0; i < HH / 4; ++i) {
        float4 v = row[i];
        uint2 p;
        unsigned short e0 = f2bf((v.x - m) * inv * g[4 * i + 0] + b[4 * i + 0]);
        unsigned short e1 = f2bf((v.y - m) * inv * g[4 * i + 1] + b[4 * i + 1]);
        unsigned short e2 = f2bf((v.z - m) * inv * g[4 * i + 2] + b[4 * i + 2]);
        unsigned short e3 = f2bf((v.w - m) * inv * g[4 * i + 3] + b[4 * i + 3]);
        p.x = (unsigned int)e0 | ((unsigned int)e1 << 16);
        p.y = (unsigned int)e2 | ((unsigned int)e3 << 16);
        *reinterpret_cast<uint2*>(o + 4 * i) = p;
    }
}

// ---------------------------------------------------------------------------
// Generic WMMA GEMM: out = act(A[N,128](bf16) @ W[128,ncols](bf16) + bias)
// blockDim = 2*ncols (one wave per 16-col tile), grid = N/16.
// ---------------------------------------------------------------------------
__global__ void wmma_gemm_kernel(const unsigned short* __restrict__ A,
                                 const unsigned short* __restrict__ W,
                                 const float* __restrict__ bias,
                                 float* __restrict__ outF,
                                 unsigned short* __restrict__ outBf,
                                 int ncols, int applyGelu) {
    const int row0 = blockIdx.x * 16;
    const int lane = threadIdx.x & 31;
    const int wave = threadIdx.x >> 5;
    const int n0   = wave * 16;
    const int m    = lane & 15;
    const int kh   = lane >> 4;

    v8f acc = {0.f, 0.f, 0.f, 0.f, 0.f, 0.f, 0.f, 0.f};
#pragma unroll
    for (int k0 = 0; k0 < HH; k0 += 32) {
        v16bf a = load_a_tile(A + (size_t)(row0 + m) * HH + k0 + kh * 8);
        v16bf bb = load_b_tile(W + (size_t)(k0 + kh * 8) * ncols + n0 + m, ncols);
        acc = __builtin_amdgcn_wmma_f32_16x16x32_bf16(
            false, a, false, bb, (short)0, acc, false, false);
    }
    const float bn = bias[n0 + (lane & 15)];
#pragma unroll
    for (int r = 0; r < 8; ++r) {
        int   mm  = r + 8 * kh;
        float v   = acc[r] + bn;
        if (applyGelu) v = gelu_exact(v);
        size_t idx = (size_t)(row0 + mm) * ncols + n0 + (lane & 15);
        outF[idx] = v;
        if (outBf) outBf[idx] = f2bf(v);
    }
}

// ---------------------------------------------------------------------------
// Edge scatter: agg[dst] += h[src]  (4 cols per thread, float atomics)
// ---------------------------------------------------------------------------
__global__ void scatter_kernel(const float* __restrict__ h,
                               const int* __restrict__ src,
                               const int* __restrict__ dst,
                               float* __restrict__ agg) {
    int gid = blockIdx.x * blockDim.x + threadIdx.x;
    int e = gid >> 5;
    if (e >= EE) return;
    int c = (gid & 31) * 4;
    int s = src[e];
    int d = dst[e];
    float4 v = *reinterpret_cast<const float4*>(h + (size_t)s * HH + c);
    float* a = agg + (size_t)d * HH + c;
    atomicAdd(a + 0, v.x);
    atomicAdd(a + 1, v.y);
    atomicAdd(a + 2, v.z);
    atomicAdd(a + 3, v.w);
}

__global__ void count_kernel(const int* __restrict__ dst,
                             float* __restrict__ cnt) {
    int e = blockIdx.x * blockDim.x + threadIdx.x;
    if (e < EE) atomicAdd(cnt + dst[e], 1.0f);
}

// mean = agg / max(cnt,1)  -> bf16
__global__ void mean_kernel(const float* __restrict__ agg,
                            const float* __restrict__ cnt,
                            unsigned short* __restrict__ meanbf) {
    int gid = blockIdx.x * blockDim.x + threadIdx.x;
    int node = gid >> 5;
    if (node >= NN) return;
    int c = (gid & 31) * 4;
    float inv = 1.0f / fmaxf(cnt[node], 1.0f);
    float4 v = *reinterpret_cast<const float4*>(agg + (size_t)node * HH + c);
    uint2 p;
    p.x = (unsigned int)f2bf(v.x * inv) | ((unsigned int)f2bf(v.y * inv) << 16);
    p.y = (unsigned int)f2bf(v.z * inv) | ((unsigned int)f2bf(v.w * inv) << 16);
    *reinterpret_cast<uint2*>(meanbf + (size_t)node * HH + c) = p;
}

// ---------------------------------------------------------------------------
// SAGE layer (fused): conv = mean@wl + bl + h@wr; h = gelu(LN(conv)) + h
// block = 256 threads (8 waves), covers 16 nodes x 128 cols. In-place on h.
// ---------------------------------------------------------------------------
__global__ __launch_bounds__(256)
void sage_layer_kernel(const unsigned short* __restrict__ meanbf,
                       unsigned short* __restrict__ hbf,
                       float* __restrict__ hF,
                       const unsigned short* __restrict__ wl,
                       const float* __restrict__ bl,
                       const unsigned short* __restrict__ wr,
                       const float* __restrict__ g,
                       const float* __restrict__ b) {
    __shared__ float sc[16][HH + 4];
    __shared__ float rmean[16];
    __shared__ float rinv[16];

    const int row0 = blockIdx.x * 16;
    const int tid  = threadIdx.x;
    const int lane = tid & 31;
    const int wave = tid >> 5;
    const int n0   = wave * 16;
    const int m    = lane & 15;
    const int kh   = lane >> 4;

    v8f acc = {0.f, 0.f, 0.f, 0.f, 0.f, 0.f, 0.f, 0.f};
#pragma unroll
    for (int k0 = 0; k0 < HH; k0 += 32) {
        // mean @ wl
        v16bf a1 = load_a_tile(meanbf + (size_t)(row0 + m) * HH + k0 + kh * 8);
        v16bf b1 = load_b_tile(wl + (size_t)(k0 + kh * 8) * HH + n0 + m, HH);
        acc = __builtin_amdgcn_wmma_f32_16x16x32_bf16(
            false, a1, false, b1, (short)0, acc, false, false);
        // h @ wr
        v16bf a2 = load_a_tile(hbf + (size_t)(row0 + m) * HH + k0 + kh * 8);
        v16bf b2 = load_b_tile(wr + (size_t)(k0 + kh * 8) * HH + n0 + m, HH);
        acc = __builtin_amdgcn_wmma_f32_16x16x32_bf16(
            false, a2, false, b2, (short)0, acc, false, false);
    }

    const float bn = bl[n0 + (lane & 15)];
#pragma unroll
    for (int r = 0; r < 8; ++r)
        sc[r + 8 * kh][n0 + (lane & 15)] = acc[r] + bn;
    __syncthreads();

    if (tid < 16) {
        float s = 0.f;
#pragma unroll 8
        for (int c = 0; c < HH; ++c) s += sc[tid][c];
        float mu = s * (1.0f / HH);
        float vv = 0.f;
#pragma unroll 8
        for (int c = 0; c < HH; ++c) {
            float d = sc[tid][c] - mu;
            vv += d * d;
        }
        rmean[tid] = mu;
        rinv[tid]  = rsqrtf(vv * (1.0f / HH) + 1e-5f);
    }
    __syncthreads();

#pragma unroll
    for (int j = 0; j < 8; ++j) {
        int idx = tid + j * 256;          // 0..2047
        int mm  = idx >> 7;
        int c   = idx & 127;
        float v = (sc[mm][c] - rmean[mm]) * rinv[mm] * g[c] + b[c];
        v = gelu_exact(v);
        size_t gi = (size_t)(row0 + mm) * HH + c;
        float o = v + hF[gi];             // residual
        hF[gi]  = o;
        hbf[gi] = f2bf(o);
    }
}

// ---------------------------------------------------------------------------
// Host launcher
// ---------------------------------------------------------------------------
extern "C" void kernel_launch(void* const* d_in, const int* in_sizes, int n_in,
                              void* d_out, int out_size, void* d_ws, size_t ws_size,
                              hipStream_t stream) {
    (void)in_sizes; (void)n_in; (void)out_size; (void)ws_size;

    const float* x          = (const float*)d_in[0];
    const int*   edge_index = (const int*)d_in[1];
    const float* in_norm_g  = (const float*)d_in[2];
    const float* in_norm_b  = (const float*)d_in[3];
    const float* in_proj_w  = (const float*)d_in[4];
    const float* in_proj_b  = (const float*)d_in[5];
    const float* lin_l_w    = (const float*)d_in[6];
    const float* lin_l_b    = (const float*)d_in[7];
    const float* lin_r_w    = (const float*)d_in[8];
    const float* norm_g     = (const float*)d_in[9];
    const float* norm_b     = (const float*)d_in[10];
    const float* out_norm_g = (const float*)d_in[11];
    const float* out_norm_b = (const float*)d_in[12];
    const float* out_proj_w = (const float*)d_in[13];
    const float* out_proj_b = (const float*)d_in[14];
    float* out = (float*)d_out;

    const int* src = edge_index;        // row 0
    const int* dst = edge_index + EE;   // row 1

    // ---- workspace carving (256B aligned) ----
    char* ws = (char*)d_ws;
    size_t off = 0;
    auto carve = [&](size_t bytes) -> char* {
        char* p = ws + off;
        off += (bytes + 255) & ~(size_t)255;
        return p;
    };
    float*          hF     = (float*)         carve((size_t)NN * HH * 4);
    float*          agg    = (float*)         carve((size_t)NN * HH * 4);
    unsigned short* hBf    = (unsigned short*)carve((size_t)NN * HH * 2);
    unsigned short* meanbf = (unsigned short*)carve((size_t)NN * HH * 2);
    unsigned short* lnbuf  = (unsigned short*)carve((size_t)NN * HH * 2);
    float*          cnt    = (float*)         carve((size_t)NN * 4);
    unsigned short* wpB    = (unsigned short*)carve((size_t)DD * HH * 2);
    unsigned short* wlB    = (unsigned short*)carve((size_t)LL * HH * HH * 2);
    unsigned short* wrB    = (unsigned short*)carve((size_t)LL * HH * HH * 2);
    unsigned short* woB    = (unsigned short*)carve((size_t)HH * OO * 2);

    // ---- convert weights to bf16 ----
    cvt_bf16_kernel<<<(DD * HH + 255) / 256, 256, 0, stream>>>(in_proj_w, wpB, DD * HH);
    cvt_bf16_kernel<<<(LL * HH * HH + 255) / 256, 256, 0, stream>>>(lin_l_w, wlB, LL * HH * HH);
    cvt_bf16_kernel<<<(LL * HH * HH + 255) / 256, 256, 0, stream>>>(lin_r_w, wrB, LL * HH * HH);
    cvt_bf16_kernel<<<(HH * OO + 255) / 256, 256, 0, stream>>>(out_proj_w, woB, HH * OO);

    // ---- input LN + projection + GELU ----
    ln_to_bf16_kernel<<<(NN + 255) / 256, 256, 0, stream>>>(x, in_norm_g, in_norm_b, lnbuf);
    wmma_gemm_kernel<<<NN / 16, 2 * HH, 0, stream>>>(
        lnbuf, wpB, in_proj_b, hF, hBf, HH, /*gelu=*/1);

    // ---- SAGE layers ----
    for (int i = 0; i < LL; ++i) {
        hipMemsetAsync(agg, 0, (size_t)NN * HH * 4, stream);
        hipMemsetAsync(cnt, 0, (size_t)NN * 4, stream);
        scatter_kernel<<<(EE * 32 + 255) / 256, 256, 0, stream>>>(hF, src, dst, agg);
        count_kernel<<<(EE + 255) / 256, 256, 0, stream>>>(dst, cnt);
        mean_kernel<<<(NN * 32 + 255) / 256, 256, 0, stream>>>(agg, cnt, meanbf);
        sage_layer_kernel<<<NN / 16, 256, 0, stream>>>(
            meanbf, hBf, hF,
            wlB + (size_t)i * HH * HH, lin_l_b + (size_t)i * HH,
            wrB + (size_t)i * HH * HH,
            norm_g + (size_t)i * HH, norm_b + (size_t)i * HH);
    }

    // ---- output LN + projection ----
    ln_to_bf16_kernel<<<(NN + 255) / 256, 256, 0, stream>>>(hF, out_norm_g, out_norm_b, lnbuf);
    wmma_gemm_kernel<<<NN / 16, 2 * OO, 0, stream>>>(
        lnbuf, woB, out_proj_b, out, (unsigned short*)nullptr, OO, /*gelu=*/0);
}
